// HyperGraphStructureLearning_78486232367517
// MI455X (gfx1250) — compile-verified, matrix operand order checked by reading, not run on hardware
//
#include <hip/hip_runtime.h>
#include <hip/hip_bf16.h>
#include <math.h>

// ---------------------------------------------------------------------------
// HyperGraph structure learning forward pass for gfx1250 (MI455X).
// All GEMM work runs through v_wmma_f32_16x16x32_bf16 (bf16 in, f32 accum).
// B fragments are batch-loaded per 16x16 tile so the WMMA chain issues with a
// single s_wait_loadcnt, letting loads of tile n+1 overlap WMMAs of tile n.
// ---------------------------------------------------------------------------

static constexpr int kN = 32768;   // nodes
static constexpr int kM = 8192;    // hyperedges
static constexpr int kD = 128;     // feature dim

typedef __attribute__((ext_vector_type(16))) __bf16 v16bf;
typedef __attribute__((ext_vector_type(8)))  __bf16 v8bf;
typedef __attribute__((ext_vector_type(8)))  float  v8f;

// --------------------------- fragment loading ------------------------------
// A operand (16x32 bf16): lane&15 selects M row; lane bit4 selects the K
// phase. Per lane: halves[0..7] = K k0..k0+7, halves[8..15] = K k0+16..k0+23
// with k0 = kbase + (lane>=16 ? 8 : 0).  B operand assumed symmetric
// (lane&15 = N column of the transposed weight, same K striping), so both
// operands load as two contiguous 16-byte reads.
__device__ inline v16bf load_frag(const __bf16* base, int kbase) {
  const int lane = threadIdx.x & 31;
  const int k0 = kbase + ((lane & 16) ? 8 : 0);
  v8bf lo = *(const v8bf*)(base + k0);
  v8bf hi = *(const v8bf*)(base + k0 + 16);
  v16bf f;
#pragma unroll
  for (int i = 0; i < 8; ++i) { f[i] = lo[i]; f[i + 8] = hi[i]; }
  return f;
}

// Preload all A fragments (row data) once per row-tile; they are reused for
// every output column tile.
template<int NIN>
__device__ inline void load_a_frags(const __bf16* const* rowp, v16bf* afr) {
#pragma unroll
  for (int i = 0; i < NIN; ++i)
#pragma unroll
    for (int ks = 0; ks < 4; ++ks)
      afr[i * 4 + ks] = load_frag(rowp[i], ks * 32);
}

// 16x16 f32 tile = sum over NIN inputs (each K=128) of A_i x B_i.
// wrow points at the packed-transposed weight row for this lane's column:
// WT[col][NIN*128] (K concatenated across inputs).  All B fragments are
// loaded first (single wait), then the WMMA chain runs back-to-back.
template<int NIN>
__device__ inline v8f mma_tile(const v16bf* afr, const __bf16* wrow) {
  v16bf bfr[NIN * 4];
#pragma unroll
  for (int i = 0; i < NIN; ++i)
#pragma unroll
    for (int ks = 0; ks < 4; ++ks)
      bfr[i * 4 + ks] = load_frag(wrow, i * 128 + ks * 32);
  v8f c = {};
#pragma unroll
  for (int i = 0; i < NIN; ++i)
#pragma unroll
    for (int ks = 0; ks < 4; ++ks)
      c = __builtin_amdgcn_wmma_f32_16x16x32_bf16(false, afr[i * 4 + ks], false,
                                                  bfr[i * 4 + ks],
                                                  (short)0, c, false, false);
  return c;
}

// ------------------------------ small utils --------------------------------
__device__ inline void atomicMaxF(float* addr, float val) {
  unsigned* u = (unsigned*)addr;
  unsigned old = *u;
  while (__uint_as_float(old) < val) {
    unsigned prev = atomicCAS(u, old, __float_as_uint(val));
    if (prev == old) break;
    old = prev;
  }
}

__global__ void fill_f32(float* p, float v, size_t n) {
  size_t i = (size_t)blockIdx.x * blockDim.x + threadIdx.x;
  if (i < n) p[i] = v;
}

__global__ void cvt_bf16(const float* __restrict__ s, __bf16* __restrict__ d, size_t n) {
  size_t i = (size_t)blockIdx.x * blockDim.x + threadIdx.x;
  if (i < n) d[i] = (__bf16)s[i];
}

// src: f32 [K][Nc] row-major  ->  dst bf16 [Nc][kstride], entry at koff+k.
__global__ void transpose_w(const float* __restrict__ src, __bf16* __restrict__ dst,
                            int K, int Nc, int kstride, int koff) {
  int gid = blockIdx.x * blockDim.x + threadIdx.x;
  if (gid >= K * Nc) return;
  int k = gid / Nc, n = gid % Nc;
  dst[(size_t)n * kstride + koff + k] = (__bf16)src[gid];
}

__global__ void sum_bias(const float* a, const float* b, const float* c,
                         float* d, int n) {
  int i = blockIdx.x * blockDim.x + threadIdx.x;
  if (i < n) d[i] = a[i] + b[i] + (c ? c[i] : 0.f);
}

// ------------------------------ bridge build -------------------------------
__global__ void bridge_accum(const float* __restrict__ ef, const int* __restrict__ te,
                             const int* __restrict__ tp, float* __restrict__ br, int T) {
  int gid = blockIdx.x * blockDim.x + threadIdx.x;
  int t = gid >> 5;
  if (t >= T) return;
  int d = (gid & 31) * 4;
  const float4 v = *(const float4*)(ef + (size_t)te[t] * kD + d);
  float* dst = br + (size_t)tp[t] * kD + d;
  atomicAdd(dst + 0, v.x); atomicAdd(dst + 1, v.y);
  atomicAdd(dst + 2, v.z); atomicAdd(dst + 3, v.w);
}

__global__ void bridge_fin(const float* __restrict__ br, const float* __restrict__ cnt,
                           __bf16* __restrict__ out, int P) {
  int gid = blockIdx.x * blockDim.x + threadIdx.x;
  int p = gid >> 5;
  if (p >= P) return;
  int d = (gid & 31) * 4;
  float inv = 1.0f / cnt[p];
  const float4 v = *(const float4*)(br + (size_t)p * kD + d);
  __bf16* o = out + (size_t)p * kD + d;
  o[0] = (__bf16)(v.x * inv); o[1] = (__bf16)(v.y * inv);
  o[2] = (__bf16)(v.z * inv); o[3] = (__bf16)(v.w * inv);
}

// ------------------------- attention logit kernel --------------------------
// hid = gelu(sum_i gather_i(f_i)@W_i + bias); logits[p,nh] = hid . a[nh];
// also atomic segment max for the scatter softmax.
template<int NIN>
__global__ void __launch_bounds__(128)
logits_kernel(const __bf16* __restrict__ f0, const int* __restrict__ i0,
              const __bf16* __restrict__ f1, const int* __restrict__ i1,
              const __bf16* __restrict__ f2, const int* __restrict__ i2,
              const __bf16* __restrict__ WT, const float* __restrict__ bias,
              const float* __restrict__ avec, const int* __restrict__ seg,
              float* __restrict__ logits, float* __restrict__ segmax, int P) {
  const int lane = threadIdx.x & 31;
  const int tile = blockIdx.x * 4 + (threadIdx.x >> 5);
  const int r0 = tile * 16;
  if (r0 >= P) return;
  const int rl = r0 + (lane & 15);
  const int rc = rl < P ? rl : P - 1;
  const __bf16* rowp[NIN];
  rowp[0] = f0 + (size_t)(i0 ? i0[rc] : rc) * kD;
  if (NIN > 1) rowp[1] = f1 + (size_t)(i1 ? i1[rc] : rc) * kD;
  if (NIN > 2) rowp[2] = f2 + (size_t)(i2 ? i2[rc] : rc) * kD;
  v16bf afr[NIN * 4];
  load_a_frags<NIN>(rowp, afr);

  float la0[8], la1[8];
#pragma unroll
  for (int v = 0; v < 8; ++v) { la0[v] = 0.f; la1[v] = 0.f; }

  for (int nt = 0; nt < 16; ++nt) {         // 256 hidden columns
    const int col = nt * 16 + (lane & 15);
    if (nt + 1 < 16)
      __builtin_prefetch(WT + (size_t)(col + 16) * (NIN * 128), 0, 3);
    v8f c = mma_tile<NIN>(afr, WT + (size_t)col * (NIN * 128));
    const float b = bias[col];
    const float wv = avec[col];             // ai flat: col = nh*128+h
    const int nh = col >> 7;                // uniform per nt
#pragma unroll
    for (int v = 0; v < 8; ++v) {
      float x = c[v] + b;
      float g = 0.5f * x * (1.0f + erff(x * 0.70710678118f)) * wv;
      if (nh == 0) la0[v] += g; else la1[v] += g;
    }
  }
  // reduce across the 16 lanes of each half (rows v / v+8)
#pragma unroll
  for (int m = 1; m < 16; m <<= 1) {
#pragma unroll
    for (int v = 0; v < 8; ++v) {
      la0[v] += __shfl_xor(la0[v], m, 32);
      la1[v] += __shfl_xor(la1[v], m, 32);
    }
  }
  if ((lane & 15) == 0) {
    const int mbase = (lane >> 4) * 8;
#pragma unroll
    for (int v = 0; v < 8; ++v) {
      int rr = r0 + mbase + v;
      if (rr < P) {
        int s = seg[rr];
        logits[(size_t)rr * 2 + 0] = la0[v];
        logits[(size_t)rr * 2 + 1] = la1[v];
        atomicMaxF(&segmax[(size_t)s * 2 + 0], la0[v]);
        atomicMaxF(&segmax[(size_t)s * 2 + 1], la1[v]);
      }
    }
  }
}

// logits -> exp(logit - segmax) in place, atomic segment sum.
__global__ void ex_kernel(float* __restrict__ lg, const float* __restrict__ segmax,
                          const int* __restrict__ seg, float* __restrict__ segsum, int P) {
  int gid = blockIdx.x * blockDim.x + threadIdx.x;
  if (gid >= P * 2) return;
  int p = gid >> 1, nh = gid & 1;
  float m = segmax[(size_t)seg[p] * 2 + nh];
  if (m < -1e37f) m = 0.f;                 // mirrors where(isfinite)
  float e = expf(lg[gid] - m);
  lg[gid] = e;
  atomicAdd(&segsum[(size_t)seg[p] * 2 + nh], e);
}

// msg = (concat inputs @ W + b) * softmax, atomic scatter into acc[seg].
template<int NIN>
__global__ void __launch_bounds__(128)
msg_kernel(const __bf16* __restrict__ f0, const int* __restrict__ i0,
           const __bf16* __restrict__ f1, const int* __restrict__ i1,
           const __bf16* __restrict__ WT, const float* __restrict__ bias,
           const float* __restrict__ ex, const float* __restrict__ segsum,
           const int* __restrict__ seg, float* __restrict__ acc, int P) {
  const int lane = threadIdx.x & 31;
  const int tile = blockIdx.x * 4 + (threadIdx.x >> 5);
  const int r0 = tile * 16;
  if (r0 >= P) return;
  const int rl = r0 + (lane & 15);
  const int rc = rl < P ? rl : P - 1;
  const __bf16* rowp[NIN];
  rowp[0] = f0 + (size_t)(i0 ? i0[rc] : rc) * kD;
  if (NIN > 1) rowp[1] = f1 + (size_t)(i1 ? i1[rc] : rc) * kD;
  v16bf afr[NIN * 4];
  load_a_frags<NIN>(rowp, afr);

  const int mbase = (lane >> 4) * 8;
  int tgt[8]; float sc0[8], sc1[8];
#pragma unroll
  for (int v = 0; v < 8; ++v) {
    int rr = r0 + mbase + v;
    int rrc = rr < P ? rr : P - 1;
    int s = seg[rrc];
    tgt[v] = s;
    sc0[v] = ex[(size_t)rrc * 2 + 0] / segsum[(size_t)s * 2 + 0];
    sc1[v] = ex[(size_t)rrc * 2 + 1] / segsum[(size_t)s * 2 + 1];
  }
  for (int nt = 0; nt < 8; ++nt) {          // 128 output columns
    const int col = nt * 16 + (lane & 15);
    if (nt + 1 < 8)
      __builtin_prefetch(WT + (size_t)(col + 16) * (NIN * 128), 0, 3);
    v8f c = mma_tile<NIN>(afr, WT + (size_t)col * (NIN * 128));
    const float b = bias[col];
    const int nh = col >> 6;                // HD=64: nh = col/64 (uniform)
#pragma unroll
    for (int v = 0; v < 8; ++v) {
      int rr = r0 + mbase + v;
      if (rr < P) {
        float s = (nh == 0) ? sc0[v] : sc1[v];
        atomicAdd(&acc[(size_t)tgt[v] * kD + col], (c[v] + b) * s);
      }
    }
  }
}

// out = acc1@Woi + acc2@Woe + (boi+boe), dense rows.
__global__ void __launch_bounds__(128)
out_kernel(const __bf16* __restrict__ f0, const __bf16* __restrict__ f1,
           const __bf16* __restrict__ WT, const float* __restrict__ bias,
           float* __restrict__ out, int Nrows) {
  const int lane = threadIdx.x & 31;
  const int tile = blockIdx.x * 4 + (threadIdx.x >> 5);
  const int r0 = tile * 16;
  if (r0 >= Nrows) return;
  const int rl = r0 + (lane & 15);
  const int rc = rl < Nrows ? rl : Nrows - 1;
  const __bf16* rowp[2] = { f0 + (size_t)rc * kD, f1 + (size_t)rc * kD };
  v16bf afr[8];
  load_a_frags<2>(rowp, afr);
  const int mbase = (lane >> 4) * 8;
  for (int nt = 0; nt < 8; ++nt) {
    const int col = nt * 16 + (lane & 15);
    if (nt + 1 < 8)
      __builtin_prefetch(WT + (size_t)(col + 16) * 256, 0, 3);
    v8f c = mma_tile<2>(afr, WT + (size_t)col * 256);
    const float b = bias[col];
#pragma unroll
    for (int v = 0; v < 8; ++v) {
      int rr = r0 + mbase + v;
      if (rr < Nrows) out[(size_t)rr * kD + col] = c[v] + b;
    }
  }
}

// ------------------------------- launcher ----------------------------------
extern "C" void kernel_launch(void* const* d_in, const int* in_sizes, int n_in,
                              void* d_out, int out_size, void* d_ws, size_t ws_size,
                              hipStream_t stream) {
  const float* node_feat = (const float*)d_in[0];
  const float* edge_feat = (const float*)d_in[1];
  const int*   inc_node  = (const int*)d_in[2];
  const int*   inc_edge  = (const int*)d_in[3];
  const int*   pair_src  = (const int*)d_in[4];
  const int*   pair_tgt  = (const int*)d_in[5];
  const float* pair_cnt  = (const float*)d_in[6];
  const int*   tri_pair  = (const int*)d_in[7];
  const int*   tri_edge  = (const int*)d_in[8];
  const float* Wsi = (const float*)d_in[9];  const float* bsi = (const float*)d_in[10];
  const float* Wti = (const float*)d_in[11]; const float* bti = (const float*)d_in[12];
  const float* Wbi = (const float*)d_in[13]; const float* bbi = (const float*)d_in[14];
  const float* ai  = (const float*)d_in[15];
  const float* Wmi = (const float*)d_in[16]; const float* bmi = (const float*)d_in[17];
  const float* Woi = (const float*)d_in[18]; const float* boi = (const float*)d_in[19];
  const float* Wse = (const float*)d_in[20]; const float* bse = (const float*)d_in[21];
  const float* Wte = (const float*)d_in[22]; const float* bte = (const float*)d_in[23];
  const float* ae  = (const float*)d_in[24];
  const float* Wme = (const float*)d_in[25]; const float* bme = (const float*)d_in[26];
  const float* Woe = (const float*)d_in[27]; const float* boe = (const float*)d_in[28];
  const int E = in_sizes[2];
  const int P = in_sizes[4];
  const int T = in_sizes[7];
  (void)n_in; (void)out_size; (void)ws_size;

  char* w = (char*)d_ws; size_t off = 0;
  auto carve = [&](size_t bytes) -> void* {
    void* p = w + off; off += (bytes + 255) & ~(size_t)255; return p;
  };
  __bf16* node_b  = (__bf16*)carve((size_t)kN * kD * 2);
  __bf16* edge_b  = (__bf16*)carve((size_t)kM * kD * 2);
  __bf16* WThid   = (__bf16*)carve((size_t)256 * 384 * 2);  // [Wsi|Wti|Wbi]^T
  __bf16* WTmsg   = (__bf16*)carve((size_t)128 * 256 * 2);  // Wmi^T
  __bf16* WThid2  = (__bf16*)carve((size_t)256 * 256 * 2);  // [Wse|Wte]^T
  __bf16* WTmsg2  = (__bf16*)carve((size_t)128 * 128 * 2);  // Wme^T
  __bf16* WTout   = (__bf16*)carve((size_t)128 * 256 * 2);  // [Woi|Woe]^T
  float* bias_hid  = (float*)carve(256 * 4);
  float* bias_hid2 = (float*)carve(256 * 4);
  float* bias_out  = (float*)carve(128 * 4);
  float*  bridge_a = (float*)carve((size_t)P * kD * 4);
  __bf16* bridge_b = (__bf16*)carve((size_t)P * kD * 2);
  float* lg1   = (float*)carve((size_t)P * 2 * 4);
  float* lg2   = (float*)carve((size_t)E * 2 * 4);
  float* nmax1 = (float*)carve((size_t)kN * 2 * 4);
  float* nsum1 = (float*)carve((size_t)kN * 2 * 4);
  float* nmax2 = (float*)carve((size_t)kN * 2 * 4);
  float* nsum2 = (float*)carve((size_t)kN * 2 * 4);
  float* acc1  = (float*)carve((size_t)kN * kD * 4);
  float* acc2  = (float*)carve((size_t)kN * kD * 4);
  __bf16* acc1b = (__bf16*)carve((size_t)kN * kD * 2);
  __bf16* acc2b = (__bf16*)carve((size_t)kN * kD * 2);

  const int TB = 256;
  auto nb = [&](size_t n) { return (unsigned)((n + TB - 1) / TB); };

  // init / precision staging
  fill_f32<<<nb((size_t)P * kD), TB, 0, stream>>>(bridge_a, 0.f, (size_t)P * kD);
  fill_f32<<<nb((size_t)kN * 2), TB, 0, stream>>>(nmax1, -3.4e38f, (size_t)kN * 2);
  fill_f32<<<nb((size_t)kN * 2), TB, 0, stream>>>(nsum1, 0.f, (size_t)kN * 2);
  fill_f32<<<nb((size_t)kN * 2), TB, 0, stream>>>(nmax2, -3.4e38f, (size_t)kN * 2);
  fill_f32<<<nb((size_t)kN * 2), TB, 0, stream>>>(nsum2, 0.f, (size_t)kN * 2);
  fill_f32<<<nb((size_t)kN * kD), TB, 0, stream>>>(acc1, 0.f, (size_t)kN * kD);
  fill_f32<<<nb((size_t)kN * kD), TB, 0, stream>>>(acc2, 0.f, (size_t)kN * kD);
  cvt_bf16<<<nb((size_t)kN * kD), TB, 0, stream>>>(node_feat, node_b, (size_t)kN * kD);
  cvt_bf16<<<nb((size_t)kM * kD), TB, 0, stream>>>(edge_feat, edge_b, (size_t)kM * kD);

  // weight packing (transposed bf16, K concatenated per multi-input GEMM)
  transpose_w<<<nb(128 * 256), TB, 0, stream>>>(Wsi, WThid, 128, 256, 384, 0);
  transpose_w<<<nb(128 * 256), TB, 0, stream>>>(Wti, WThid, 128, 256, 384, 128);
  transpose_w<<<nb(128 * 256), TB, 0, stream>>>(Wbi, WThid, 128, 256, 384, 256);
  transpose_w<<<nb(256 * 128), TB, 0, stream>>>(Wmi, WTmsg, 256, 128, 256, 0);
  transpose_w<<<nb(128 * 256), TB, 0, stream>>>(Wse, WThid2, 128, 256, 256, 0);
  transpose_w<<<nb(128 * 256), TB, 0, stream>>>(Wte, WThid2, 128, 256, 256, 128);
  transpose_w<<<nb(128 * 128), TB, 0, stream>>>(Wme, WTmsg2, 128, 128, 128, 0);
  transpose_w<<<nb(128 * 128), TB, 0, stream>>>(Woi, WTout, 128, 128, 256, 0);
  transpose_w<<<nb(128 * 128), TB, 0, stream>>>(Woe, WTout, 128, 128, 256, 128);
  sum_bias<<<1, 256, 0, stream>>>(bsi, bti, bbi, bias_hid, 256);
  sum_bias<<<1, 256, 0, stream>>>(bse, bte, nullptr, bias_hid2, 256);
  sum_bias<<<1, 256, 0, stream>>>(boi, boe, nullptr, bias_out, 128);

  // bridge features (mean-coalesced over triples)
  bridge_accum<<<nb((size_t)T * 32), TB, 0, stream>>>(edge_feat, tri_edge, tri_pair, bridge_a, T);
  bridge_fin<<<nb((size_t)P * 32), TB, 0, stream>>>(bridge_a, pair_cnt, bridge_b, P);

  // ---- IntraRank (pair) branch ----
  {
    unsigned tiles = (unsigned)((P + 15) / 16), blk = (tiles + 3) / 4;
    logits_kernel<3><<<blk, 128, 0, stream>>>(node_b, pair_src, node_b, pair_tgt,
        bridge_b, (const int*)nullptr, WThid, bias_hid, ai, pair_tgt, lg1, nmax1, P);
    ex_kernel<<<nb((size_t)P * 2), TB, 0, stream>>>(lg1, nmax1, pair_tgt, nsum1, P);
    msg_kernel<2><<<blk, 128, 0, stream>>>(node_b, pair_src, bridge_b, (const int*)nullptr,
        WTmsg, bmi, lg1, nsum1, pair_tgt, acc1, P);
  }
  // ---- InterRank (edge->node) branch ----
  {
    unsigned tiles = (unsigned)((E + 15) / 16), blk = (tiles + 3) / 4;
    logits_kernel<2><<<blk, 128, 0, stream>>>(edge_b, inc_edge, node_b, inc_node,
        (const __bf16*)nullptr, (const int*)nullptr, WThid2, bias_hid2, ae, inc_node,
        lg2, nmax2, E);
    ex_kernel<<<nb((size_t)E * 2), TB, 0, stream>>>(lg2, nmax2, inc_node, nsum2, E);
    msg_kernel<1><<<blk, 128, 0, stream>>>(edge_b, inc_edge, (const __bf16*)nullptr,
        (const int*)nullptr, WTmsg2, bme, lg2, nsum2, inc_node, acc2, E);
  }
  // ---- final projection ----
  cvt_bf16<<<nb((size_t)kN * kD), TB, 0, stream>>>(acc1, acc1b, (size_t)kN * kD);
  cvt_bf16<<<nb((size_t)kN * kD), TB, 0, stream>>>(acc2, acc2b, (size_t)kN * kD);
  {
    unsigned tiles = kN / 16, blk = (tiles + 3) / 4;
    out_kernel<<<blk, 128, 0, stream>>>(acc1b, acc2b, WTout, bias_out, (float*)d_out, kN);
  }
}
